// DownsampleLayer_19430432047360
// MI455X (gfx1250) — matrix operand order repaired and checked by reading, not compiled, not run on hardware
//
#include <hip/hip_runtime.h>
#include <cstdint>

// ---------------- problem constants ----------------
#define BB 8
#define NN 8192
#define DIM_ 128
#define MM 2048
#define KK 16
#define HEADS_ 4
#define DHEAD_ 32
#define NGRP 8
#define STEP_ 0.1f
#define LAMBD_ 0.5f

typedef _Float16 h16 __attribute__((ext_vector_type(16)));
typedef float f8 __attribute__((ext_vector_type(8)));

// WMMA 16x16x32 f16 fragment index helpers (CDNA5 ISA 7.12.2)
// A (16x32, 16-bit): lanes 0-15 rows 0-15 K={0..7,16..23}; lanes 16-31 K={8..15,24..31}
__device__ __forceinline__ int wmma_ak(int j, int hi) { return (j < 8) ? (8 * hi + j) : (8 + 8 * hi + j); }
// B (32x16, 16-bit): lanes 0-15 cols K={0..7,16..23}; lanes 16-31 K={8..15,24..31}
__device__ __forceinline__ int wmma_bk(int j, int hi) { return (j < 8) ? (8 * hi + j) : (8 + 8 * hi + j); }
// C/D (16x16 f32): vgpr r -> row r + 8*hi, col = lane&15

// ---------------- FPS ----------------
__global__ __launch_bounds__(1024) void k_fps(const float* __restrict__ xyz, int* __restrict__ sidx) {
    __shared__ float dist[NN];
    __shared__ float rv[1024];
    __shared__ int ri[1024];
    const int b = blockIdx.x, tid = threadIdx.x;
    const float* xb = xyz + (size_t)b * 3 * NN;
    for (int n = tid; n < NN; n += 1024) dist[n] = 1e10f;
    __syncthreads();
    int far = 0;
    for (int m = 0; m < MM; ++m) {
        if (tid == 0) sidx[(size_t)b * MM + m] = far;
        float cx = xb[far], cy = xb[NN + far], cz = xb[2 * NN + far];
        float bv = -1.0f; int bi = 0;
        for (int n = tid; n < NN; n += 1024) {
            float dx = xb[n] - cx, dy = xb[NN + n] - cy, dz = xb[2 * NN + n] - cz;
            float d2 = dx * dx + dy * dy + dz * dz;
            float nd = fminf(dist[n], d2);
            dist[n] = nd;
            if (nd > bv) { bv = nd; bi = n; }
        }
        rv[tid] = bv; ri[tid] = bi;
        __syncthreads();
        for (int s = 512; s > 0; s >>= 1) {
            if (tid < s) {
                float ov = rv[tid + s]; int oi = ri[tid + s];
                if (ov > rv[tid] || (ov == rv[tid] && oi < ri[tid])) { rv[tid] = ov; ri[tid] = oi; }
            }
            __syncthreads();
        }
        far = ri[0];
        __syncthreads();
    }
}

// ---------------- gather sampled xyz -> out0 (B,3,M) ----------------
__global__ void k_gather_sxyz(const float* __restrict__ xyz, const int* __restrict__ sidx,
                              float* __restrict__ out0) {
    int i = blockIdx.x * 256 + threadIdx.x;
    if (i >= BB * 3 * MM) return;
    int b = i / (3 * MM), r = i % (3 * MM);
    int c = r / MM, m = r % MM;
    out0[i] = xyz[(size_t)b * 3 * NN + (size_t)c * NN + sidx[(size_t)b * MM + m]];
}

// ---------------- o2s: nearest sampled center for each point ----------------
__global__ __launch_bounds__(256) void k_o2s(const float* __restrict__ xyz, const float* __restrict__ sx,
                                             int* __restrict__ o2s) {
    __shared__ float s[3 * MM];
    const int b = blockIdx.y;
    const float* sb = sx + (size_t)b * 3 * MM;
    for (int t = threadIdx.x; t < 3 * MM; t += 256) s[t] = sb[t];
    __syncthreads();
    int n = blockIdx.x * 256 + threadIdx.x;
    const float* xb = xyz + (size_t)b * 3 * NN;
    float px = xb[n], py = xb[NN + n], pz = xb[2 * NN + n];
    float best = 3.402823466e38f; int bi = 0;
    for (int m = 0; m < MM; ++m) {
        float dx = px - s[m], dy = py - s[MM + m], dz = pz - s[2 * MM + m];
        float d = dx * dx + dy * dy + dz * dz;
        if (d < best) { best = d; bi = m; }
    }
    o2s[(size_t)b * NN + n] = bi;
}

// ---------------- kNN (K=16) of each sampled point among all N ----------------
__global__ __launch_bounds__(256) void k_knn(const float* __restrict__ xyz, const float* __restrict__ sx,
                                             int* __restrict__ knn) {
    int q = blockIdx.x * 256 + threadIdx.x;
    int b = blockIdx.y;
    const float* xb = xyz + (size_t)b * 3 * NN;
    const float* sb = sx + (size_t)b * 3 * MM;
    float qx = sb[q], qy = sb[MM + q], qz = sb[2 * MM + q];
    float bd[KK]; int bi[KK];
    for (int k = 0; k < KK; ++k) { bd[k] = 3.402823466e38f; bi[k] = 0; }
    for (int n = 0; n < NN; ++n) {
        float dx = qx - xb[n], dy = qy - xb[NN + n], dz = qz - xb[2 * NN + n];
        float d = dx * dx + dy * dy + dz * dz;
        if (d < bd[KK - 1]) {
            int pos = KK - 1;
            while (pos > 0 && d < bd[pos - 1]) { bd[pos] = bd[pos - 1]; bi[pos] = bi[pos - 1]; --pos; }
            bd[pos] = d; bi[pos] = n;
        }
    }
    for (int k = 0; k < KK; ++k) knn[((size_t)b * MM + q) * KK + k] = bi[k];
}

// ---------------- misc small kernels ----------------
__global__ void k_zero(float* p, int n) { int i = blockIdx.x * 256 + threadIdx.x; if (i < n) p[i] = 0.f; }

__global__ void k_scatter_dn(const int* __restrict__ o2s, float* __restrict__ dn) {
    int i = blockIdx.x * 256 + threadIdx.x;
    if (i >= BB * NN) return;
    int b = i / NN;
    atomicAdd(&dn[(size_t)b * MM + o2s[i]], 1.0f);
}

// lps (4,M per batch) = sum_k [direction, dn2]; also dist and mask
__global__ void k_lps(const float* __restrict__ xyz, const float* __restrict__ sx,
                      const int* __restrict__ knn, const int* __restrict__ o2s,
                      float* __restrict__ lps, float* __restrict__ dist, int* __restrict__ maskb) {
    int i = blockIdx.x * 256 + threadIdx.x;
    if (i >= BB * MM) return;
    int b = i / MM, m = i % MM;
    const float* xb = xyz + (size_t)b * 3 * NN;
    const float* sb = sx + (size_t)b * 3 * MM;
    float sx0 = sb[m], sx1 = sb[MM + m], sx2 = sb[2 * MM + m];
    float a0 = 0, a1 = 0, a2 = 0, a3 = 0;
    for (int k = 0; k < KK; ++k) {
        int j = knn[((size_t)b * MM + m) * KK + k];
        float d0 = xb[j] - sx0, d1 = xb[NN + j] - sx1, d2v = xb[2 * NN + j] - sx2;
        float dd = d0 * d0 + d1 * d1 + d2v * d2v;
        float d = sqrtf(fmaxf(dd, 1e-24f));
        dist[((size_t)b * MM + m) * KK + k] = d;
        float dm = fmaxf(d, 1e-12f);
        a0 += d0 / dm; a1 += d1 / dm; a2 += d2v / dm; a3 += d;
        int real = o2s[(size_t)b * NN + j];
        maskb[((size_t)b * MM + m) * KK + k] = (real == m) ? 1 : 0;
    }
    float* lb = lps + (size_t)b * 4 * MM;
    lb[m] = a0; lb[MM + m] = a1; lb[2 * MM + m] = a2; lb[3 * MM + m] = a3;
}

__global__ void k_md(const float* __restrict__ dist, const int* __restrict__ maskb,
                     const float* __restrict__ dn, float* __restrict__ md) {
    int i = blockIdx.x * 256 + threadIdx.x;
    if (i >= BB * MM) return;
    float s = 0;
    for (int k = 0; k < KK; ++k)
        if (maskb[(size_t)i * KK + k]) s += dist[(size_t)i * KK + k];
    md[i] = s / dn[i];
}

__global__ void k_sq(const float* __restrict__ pe, float* __restrict__ sq) {
    int i = blockIdx.x * 256 + threadIdx.x;
    if (i >= BB * MM) return;
    int b = i / MM, m = i % MM;
    const float* pb = pe + (size_t)b * DIM_ * MM;
    float s = 0;
    for (int c = 0; c < DIM_; ++c) { float v = pb[(size_t)c * MM + m]; s += v * v; }
    sq[i] = s;
}

__global__ void k_ista(const float* __restrict__ ae, const float* __restrict__ t2,
                       const float* __restrict__ t3, float* __restrict__ ff, int n) {
    int i = blockIdx.x * 256 + threadIdx.x;
    if (i >= n) return;
    ff[i] = fmaxf(ae[i] + STEP_ * (t2[i] - t3[i]) - STEP_ * LAMBD_, 0.f);
}

// ---------------- GroupNorm (in-place) ----------------
__global__ __launch_bounds__(256) void k_gn(float* __restrict__ x, const float* __restrict__ g,
                                            const float* __restrict__ bt, int cpg, int S, int relu,
                                            size_t bstride) {
    __shared__ float r1[256], r2[256];
    __shared__ float s_mu, s_rstd;
    int grp = blockIdx.x, b = blockIdx.y, tid = threadIdx.x;
    float* base = x + (size_t)b * bstride + (size_t)grp * cpg * S;
    size_t cnt = (size_t)cpg * S;
    float s = 0.f, ss = 0.f;
    for (size_t i = tid; i < cnt; i += 256) { float v = base[i]; s += v; ss += v * v; }
    r1[tid] = s; r2[tid] = ss;
    __syncthreads();
    for (int st = 128; st > 0; st >>= 1) {
        if (tid < st) { r1[tid] += r1[tid + st]; r2[tid] += r2[tid + st]; }
        __syncthreads();
    }
    if (tid == 0) {
        float mu = r1[0] / (float)cnt;
        float var = r2[0] / (float)cnt - mu * mu;
        s_mu = mu; s_rstd = rsqrtf(fmaxf(var, 0.f) + 1e-5f);
    }
    __syncthreads();
    float mu = s_mu, rstd = s_rstd;
    for (size_t i = tid; i < cnt; i += 256) {
        int c = grp * cpg + (int)(i / (size_t)S);
        float v = (base[i] - mu) * rstd * g[c] + bt[c];
        if (relu) v = fmaxf(v, 0.f);
        base[i] = v;
    }
}

// ---------------- generic WMMA pconv GEMM: Y(128 x P) = W(128 x Cin) * X(Cin x P) ----------------
// Register-blocked: each wave computes 4 column tiles (64 points), reusing the weight A-fragment.
enum { XD = 0, XGATH = 1, XNF = 2, XPE2 = 3, XQKP = 4, XCAT3 = 5 };
enum { E_STORE = 0, E_MAXK = 1, E_ATTNK = 2 };
enum { R_NONE = 0, R_DENSE = 1, R_GATH = 2 };

template <int XM, int EPI, int RESM>
__global__ __launch_bounds__(128) void gemm_k(
    const float* __restrict__ W, const float* __restrict__ bias, float bias_scale,
    const float* __restrict__ X0, const float* __restrict__ X1, const float* __restrict__ X2,
    const int* __restrict__ I0, const int* __restrict__ MK,
    const float* __restrict__ R, const int* __restrict__ RI,
    float* __restrict__ Y, int Cin, int P, int ld0, int ld1, int Mo, float oscale) {
    __shared__ float tile[4][16][17];
    const int lane = threadIdx.x & 31;
    const int w = threadIdx.x >> 5;
    const int lo = lane & 15;
    const int hi = lane >> 4;
    const int p0 = (blockIdx.x * 4 + w) * 64;   // wave covers 4 x 16 points
    const int c0 = blockIdx.y * 16;

    f8 acc[4];
#pragma unroll
    for (int t = 0; t < 4; ++t)
#pragma unroll
        for (int r = 0; r < 8; ++r) acc[t][r] = 0.f;

    for (int k0 = 0; k0 < Cin; k0 += 32) {
        h16 aF;  // weight fragment, reused across the 4 column tiles
#pragma unroll
        for (int j = 0; j < 16; ++j) {
            int kk = k0 + wmma_ak(j, hi);
            float v = (kk < Cin) ? W[(size_t)(c0 + lo) * Cin + kk] : 0.f;
            aF[j] = (_Float16)v;
        }
#pragma unroll
        for (int t = 0; t < 4; ++t) {
            const int p = p0 + t * 16 + lo;
            h16 bF;
#pragma unroll
            for (int j = 0; j < 16; ++j) {
                int kk = k0 + wmma_bk(j, hi);
                float v = 0.f;
                if (kk < Cin) {
                    if (XM == XD) v = X0[(size_t)kk * P + p];
                    else if (XM == XGATH) v = X0[(size_t)kk * ld0 + I0[p]];
                    else if (XM == XNF) {
                        int m = p >> 4;
                        if (kk < 128) v = X0[(size_t)kk * ld0 + m];
                        else if (kk < 256) v = X0[(size_t)(kk - 128) * ld0 + I0[p]];
                        else v = X1[p];
                    } else if (XM == XPE2) {
                        int m = p >> 4;
                        v = X1[(size_t)kk * ld1 + m] - X0[(size_t)kk * ld0 + I0[p]];
                    } else if (XM == XQKP) {
                        int m = p >> 4;
                        v = X0[(size_t)kk * ld1 + m] - X1[(size_t)kk * ld0 + I0[p]] + X2[(size_t)kk * P + p];
                    } else { // XCAT3
                        v = (kk < 128) ? X0[(size_t)kk * P + p]
                            : (kk < 256) ? X1[(size_t)(kk - 128) * P + p]
                                         : X2[(size_t)(kk - 256) * P + p];
                    }
                }
                bF[j] = (_Float16)v;
            }
            acc[t] = __builtin_amdgcn_wmma_f32_16x16x32_f16(false, aF, false, bF, (short)0, acc[t], false, false);
        }
    }

    if (EPI == E_STORE) {
#pragma unroll
        for (int t = 0; t < 4; ++t) {
            const int p = p0 + t * 16 + lo;
#pragma unroll
            for (int r = 0; r < 8; ++r) {
                int ch = c0 + r + 8 * hi;
                float v = acc[t][r] + bias[ch] * bias_scale;
                if (RESM == R_DENSE) v += R[(size_t)ch * P + p];
                else if (RESM == R_GATH) v += R[(size_t)ch * ld0 + RI[p]];
                Y[(size_t)ch * P + p] = v;
            }
        }
    } else if (EPI == E_MAXK) {
        for (int t = 0; t < 4; ++t) {
#pragma unroll
            for (int r = 0; r < 8; ++r) tile[w][r + 8 * hi][lo] = acc[t][r];
            __syncthreads();
            if (hi == 0) {
                int row = lo;
                float mx = tile[w][row][0];
#pragma unroll
                for (int c = 1; c < 16; ++c) mx = fmaxf(mx, tile[w][row][c]);
                int m = (p0 + t * 16) >> 4;
                Y[(size_t)(c0 + row) * Mo + m] = mx + bias[c0 + row];
            }
            __syncthreads();
        }
    } else { // E_ATTNK: scale, mask, softmax over the 16 neighbors, weighted sum with (key+pos)
        for (int t = 0; t < 4; ++t) {
#pragma unroll
            for (int r = 0; r < 8; ++r) {
                int ch = c0 + r + 8 * hi;
                tile[w][r + 8 * hi][lo] = (acc[t][r] + bias[ch]) * oscale;
            }
            __syncthreads();
            if (hi == 0) {
                int ch = c0 + lo;
                int m = (p0 + t * 16) >> 4;
                const float NEGMAX = -3.402823466e38f;
                float l[16]; float mx = NEGMAX;
#pragma unroll
                for (int k = 0; k < 16; ++k) {
                    float v = MK[m * 16 + k] ? tile[w][lo][k] : NEGMAX;
                    l[k] = v; mx = fmaxf(mx, v);
                }
                float s = 0.f, o = 0.f;
#pragma unroll
                for (int k = 0; k < 16; ++k) {
                    float e = expf(l[k] - mx);
                    s += e;
                    float kp = R[(size_t)ch * ld0 + I0[m * 16 + k]] + X2[(size_t)ch * P + m * 16 + k];
                    o += e * kp;
                }
                Y[(size_t)ch * Mo + m] = o / s;
            }
            __syncthreads();
        }
    }
}

// ---------------- row-major GEMM for ISTA: Y(P x 128) = X(P x 128) * (W or W^T) ----------------
// Register-blocked: weight B-fragment hoisted, 4 X row-tiles per wave.
template <bool WT>
__global__ __launch_bounds__(128) void gemm_xw_k(const float* __restrict__ X, const float* __restrict__ W,
                                                 float* __restrict__ Y, int P) {
    const int lane = threadIdx.x & 31;
    const int w = threadIdx.x >> 5;
    const int lo = lane & 15, hi = lane >> 4;
    const int p0 = (blockIdx.x * 4 + w) * 64;
    const int o0 = blockIdx.y * 16;
    f8 acc[4];
#pragma unroll
    for (int t = 0; t < 4; ++t)
#pragma unroll
        for (int r = 0; r < 8; ++r) acc[t][r] = 0.f;
    for (int k0 = 0; k0 < 128; k0 += 32) {
        h16 bF;  // weight fragment, reused across 4 row tiles
#pragma unroll
        for (int j = 0; j < 16; ++j) {
            int kk = k0 + wmma_bk(j, hi);
            float v = WT ? W[(size_t)(o0 + lo) * 128 + kk] : W[(size_t)kk * 128 + o0 + lo];
            bF[j] = (_Float16)v;
        }
#pragma unroll
        for (int t = 0; t < 4; ++t) {
            h16 aF;
#pragma unroll
            for (int j = 0; j < 16; ++j)
                aF[j] = (_Float16)X[(size_t)(p0 + t * 16 + lo) * 128 + k0 + wmma_ak(j, hi)];
            acc[t] = __builtin_amdgcn_wmma_f32_16x16x32_f16(false, aF, false, bF, (short)0, acc[t], false, false);
        }
    }
#pragma unroll
    for (int t = 0; t < 4; ++t)
#pragma unroll
        for (int r = 0; r < 8; ++r)
            Y[(size_t)(p0 + t * 16 + r + 8 * hi) * 128 + o0 + lo] = acc[t][r];
}

// ---------------- flash attention over the M=2048 sampled points ----------------
__global__ __launch_bounds__(32) void k_attn(const float* __restrict__ wbuf, float* __restrict__ fea) {
    const int mt = blockIdx.x, h = blockIdx.y, b = blockIdx.z;
    const float* Wb = wbuf + ((size_t)b * DIM_ + h * DHEAD_) * MM;
    const int lane = threadIdx.x, lo = lane & 15, hi = lane >> 4;
    __shared__ float sS[16][17];
    __shared__ float sV[16][33];
    __shared__ float smax[16], ssum[16], scorr[16];
    const int m0 = mt * 16;
    h16 aF;
#pragma unroll
    for (int j = 0; j < 16; ++j)
        aF[j] = (_Float16)Wb[(size_t)wmma_ak(j, hi) * MM + m0 + lo];
    float racc[16];
#pragma unroll
    for (int c = 0; c < 16; ++c) racc[c] = 0.f;
    if (lane < 16) { smax[lane] = -3.402823466e38f; ssum[lane] = 0.f; }
    __syncthreads();
    const float scale = 0.17677669529663687f; // 1/sqrt(32)
    for (int jt = 0; jt < MM / 16; ++jt) {
        int c0 = jt * 16;
        h16 bF;
#pragma unroll
        for (int j = 0; j < 16; ++j)
            bF[j] = (_Float16)Wb[(size_t)wmma_bk(j, hi) * MM + c0 + lo];
        f8 cc;
#pragma unroll
        for (int r = 0; r < 8; ++r) cc[r] = 0.f;
        cc = __builtin_amdgcn_wmma_f32_16x16x32_f16(false, aF, false, bF, (short)0, cc, false, false);
        for (int t = lane; t < 512; t += 32) {
            int k = t & 15, d = t >> 4;
            sV[k][d] = Wb[(size_t)d * MM + c0 + k];
        }
#pragma unroll
        for (int r = 0; r < 8; ++r) sS[r + 8 * hi][lo] = cc[r] * scale;
        __syncthreads();
        if (lane < 16) {
            int row = lane;
            float mx = smax[row];
#pragma unroll
            for (int k = 0; k < 16; ++k) mx = fmaxf(mx, sS[row][k]);
            float corr = expf(smax[row] - mx);
            float s = ssum[row] * corr;
#pragma unroll
            for (int k = 0; k < 16; ++k) { float e = expf(sS[row][k] - mx); sS[row][k] = e; s += e; }
            smax[row] = mx; ssum[row] = s; scorr[row] = corr;
        }
        __syncthreads();
        {
            float corr = scorr[lo];
#pragma unroll
            for (int c = 0; c < 16; ++c) racc[c] *= corr;
            for (int k = 0; k < 16; ++k) {
                float pk = sS[lo][k];
#pragma unroll
                for (int c = 0; c < 16; ++c) racc[c] += pk * sV[k][hi * 16 + c];
            }
        }
        __syncthreads();
    }
    float inv = 1.0f / ssum[lo];
#pragma unroll
    for (int c = 0; c < 16; ++c)
        fea[((size_t)b * DIM_ + h * DHEAD_ + hi * 16 + c) * MM + m0 + lo] = racc[c] * inv;
}

// ---------------- feature-space kNN (get_graph_feature): WMMA inner products + top-16 ----------------
__global__ __launch_bounds__(128) void k_graphfeat(const float* __restrict__ pe, const float* __restrict__ sq,
                                                   int* __restrict__ gfidx, float* __restrict__ nrm) {
    __shared__ float sc[16 * 512];
    const int b = blockIdx.y, nt = blockIdx.x, n0 = nt * 16;
    const int lane = threadIdx.x & 31, w = threadIdx.x >> 5, lo = lane & 15, hi = lane >> 4;
    const float* peb = pe + (size_t)b * DIM_ * MM;
    const float* sqb = sq + (size_t)b * MM;
    h16 aF[4];
    for (int ch = 0; ch < 4; ++ch) {
#pragma unroll
        for (int j = 0; j < 16; ++j)
            aF[ch][j] = (_Float16)peb[(size_t)(32 * ch + wmma_ak(j, hi)) * MM + n0 + lo];
    }
    float bd[16]; int bi[16];
    if (threadIdx.x < 16) {
        for (int k = 0; k < 16; ++k) { bd[k] = -3.402823466e38f; bi[k] = 0; }
    }
    for (int pass = 0; pass < 4; ++pass) {
        __syncthreads();
        for (int ct = pass * 32 + w; ct < pass * 32 + 32; ct += 4) {
            f8 acc;
#pragma unroll
            for (int r = 0; r < 8; ++r) acc[r] = 0.f;
            for (int ch = 0; ch < 4; ++ch) {
                h16 bF;
#pragma unroll
                for (int j = 0; j < 16; ++j)
                    bF[j] = (_Float16)peb[(size_t)(32 * ch + wmma_bk(j, hi)) * MM + ct * 16 + lo];
                acc = __builtin_amdgcn_wmma_f32_16x16x32_f16(false, aF[ch], false, bF, (short)0, acc, false, false);
            }
            int col = ct * 16 + lo;
            float sqc = sqb[col];
            int lcol = (ct - pass * 32) * 16 + lo;
#pragma unroll
            for (int r = 0; r < 8; ++r)
                sc[(r + 8 * hi) * 512 + lcol] = 2.f * acc[r] - sqc;
        }
        __syncthreads();
        if (threadIdx.x < 16) {
            const float* srow = sc + threadIdx.x * 512;
            int cbase = pass * 512;
            for (int c = 0; c < 512; ++c) {
                float s = srow[c];
                if (s > bd[15]) {
                    int pos = 15;
                    while (pos > 0 && s > bd[pos - 1]) { bd[pos] = bd[pos - 1]; bi[pos] = bi[pos - 1]; --pos; }
                    bd[pos] = s; bi[pos] = cbase + c;
                }
            }
        }
    }
    if (threadIdx.x < 16) {
        int q = n0 + threadIdx.x;
        float sqq = sqb[q];
        for (int k = 0; k < 16; ++k) {
            gfidx[((size_t)b * MM + q) * KK + k] = bi[k];
            nrm[((size_t)b * MM + q) * KK + k] = sqrtf(fmaxf(sqq - bd[k], 1e-24f));
        }
    }
}

// ---------------- host driver ----------------
enum {
    pW_pre, pb_pre, pW_w, pb_w, pW_out, pb_out, pff_W, pff_gn_s, pff_gn_b,
    pW_ffh, pb_ffh, pW_prenn, pb_prenn, pW_lc1, pb_lc1, pgn_lc_s, pgn_lc_b,
    pW_lc2, pb_lc2, pW_qkv, pb_qkv, pW_p1, pb_p1, pgn_p_s, pgn_p_b,
    pW_p2, pb_p2, pW_a1, pb_a1, pgn_a_s, pgn_a_b, pW_a2, pb_a2,
    pW_s1, pb_s1, pgn_s_s, pgn_s_b, pW_s2, pb_s2, pW_post, pb_post, NPARAM
};

extern "C" void kernel_launch(void* const* d_in, const int* in_sizes, int n_in,
                              void* d_out, int out_size, void* d_ws, size_t ws_size,
                              hipStream_t stream) {
    const float* xyzs = (const float*)d_in[0];
    const float* feats = (const float*)d_in[1];
    const float* Pp[NPARAM];
    for (int i = 0; i < NPARAM; ++i) Pp[i] = (2 + i < n_in) ? (const float*)d_in[2 + i] : nullptr;

    float* out0 = (float*)d_out;                  // sampled_xyzs (B,3,M)
    float* out1 = out0 + (size_t)BB * 3 * MM;     // sampled_feats (B,128,M)
    float* dn = out1 + (size_t)BB * DIM_ * MM;    // (B,M)
    float* md = dn + (size_t)BB * MM;             // (B,M)

    char* wp = (char*)d_ws;
    auto fal = [&](size_t n) -> float* { float* r = (float*)wp; wp += ((n * 4 + 255) & ~(size_t)255); return r; };
    auto ial = [&](size_t n) -> int* { int* r = (int*)wp; wp += ((n * 4 + 255) & ~(size_t)255); return r; };

    const size_t CM = (size_t)DIM_ * MM;   // per-batch (128 x M)
    const size_t CN = (size_t)DIM_ * NN;   // per-batch (128 x N)
    const size_t MKs = (size_t)MM * KK;

    float* f_kv = fal((size_t)BB * CN);
    float* f_sam = fal((size_t)BB * CM);
    float* f_w = fal((size_t)BB * CM);
    float* f_fea = fal((size_t)BB * CM);
    float* f_ae = fal((size_t)BB * CM);
    float* f_t1 = fal((size_t)BB * CM);
    float* f_t2 = fal((size_t)BB * CM);
    float* f_t3 = fal((size_t)BB * CM);
    float* f_ff = fal((size_t)BB * CM);
    float* f_creat = fal((size_t)BB * CM);
    float* f_q = fal((size_t)BB * CM);
    float* f_pe = fal((size_t)BB * CM);
    float* f_intra = fal((size_t)BB * CM);
    float* f_fgt = fal((size_t)BB * CM);
    float* f_h3 = fal((size_t)BB * CM);
    float* f_inter = fal((size_t)BB * CM);
    float* f_lps = fal((size_t)BB * 4 * MM);
    float* f_dist = fal((size_t)BB * MKs);
    float* f_nrm = fal((size_t)BB * MKs);
    float* f_sq = fal((size_t)BB * MM);
    float* f_S1 = fal((size_t)DIM_ * MKs);
    float* f_S2 = fal((size_t)DIM_ * MKs);
    int* i_sidx = ial((size_t)BB * MM);
    int* i_o2s = ial((size_t)BB * NN);
    int* i_knn = ial((size_t)BB * MKs);
    int* i_gf = ial((size_t)BB * MKs);
    int* i_mask = ial((size_t)BB * MKs);

    const float INVSQD = 0.08838834764831845f; // 1/sqrt(128)

    // ---- indices ----
    k_fps<<<BB, 1024, 0, stream>>>(xyzs, i_sidx);
    k_gather_sxyz<<<(BB * 3 * MM) / 256, 256, 0, stream>>>(xyzs, i_sidx, out0);
    k_o2s<<<dim3(NN / 256, BB), 256, 0, stream>>>(xyzs, out0, i_o2s);
    k_knn<<<dim3(MM / 256, BB), 256, 0, stream>>>(xyzs, out0, i_knn);
    k_zero<<<(BB * MM) / 256, 256, 0, stream>>>(dn, BB * MM);
    k_scatter_dn<<<(BB * NN) / 256, 256, 0, stream>>>(i_o2s, dn);
    k_lps<<<(BB * MM) / 256, 256, 0, stream>>>(xyzs, out0, i_knn, i_o2s, f_lps, f_dist, i_mask);
    k_md<<<(BB * MM) / 256, 256, 0, stream>>>(f_dist, i_mask, dn, md);

    // ---- sam_feats, qkv, attention inputs ----
    for (int b = 0; b < BB; ++b) {
        const float* fb = feats + (size_t)b * CN;
        gemm_k<XGATH, E_STORE, R_NONE><<<dim3(MM / 256, 8), 128, 0, stream>>>(
            Pp[pW_pre], Pp[pb_pre], 1.f, fb, nullptr, nullptr, i_sidx + (size_t)b * MM, nullptr,
            nullptr, nullptr, f_sam + b * CM, DIM_, MM, NN, 0, 0, 1.f);
        gemm_k<XD, E_STORE, R_NONE><<<dim3(MM / 256, 8), 128, 0, stream>>>(
            Pp[pW_w], Pp[pb_w], 1.f, f_sam + b * CM, nullptr, nullptr, nullptr, nullptr,
            nullptr, nullptr, f_w + b * CM, DIM_, MM, 0, 0, 0, 1.f);
        gemm_k<XD, E_STORE, R_NONE><<<dim3(MM / 256, 8), 128, 0, stream>>>(
            Pp[pW_qkv], Pp[pb_qkv], 1.f, f_sam + b * CM, nullptr, nullptr, nullptr, nullptr,
            nullptr, nullptr, f_q + b * CM, DIM_, MM, 0, 0, 0, 1.f);
        gemm_k<XD, E_STORE, R_NONE><<<dim3(NN / 256, 8), 128, 0, stream>>>(
            Pp[pW_qkv], Pp[pb_qkv], 1.f, fb, nullptr, nullptr, nullptr, nullptr,
            nullptr, nullptr, f_kv + b * CN, DIM_, NN, 0, 0, 0, 1.f);
    }

    // ---- full MxM self-attention ----
    k_attn<<<dim3(MM / 16, HEADS_, BB), 32, 0, stream>>>(f_w, f_fea);
    for (int b = 0; b < BB; ++b)
        gemm_k<XD, E_STORE, R_DENSE><<<dim3(MM / 256, 8), 128, 0, stream>>>(
            Pp[pW_out], Pp[pb_out], 1.f, f_fea + b * CM, nullptr, nullptr, nullptr, nullptr,
            f_sam + b * CM, nullptr, f_ae + b * CM, DIM_, MM, 0, 0, 0, 1.f);

    // ---- ISTA feedforward (raw-reshape view: flat (B*M,128) row-major) ----
    gemm_xw_k<true><<<dim3((BB * MM) / 256, 8), 128, 0, stream>>>(f_ae, Pp[pff_W], f_t1, BB * MM);
    gemm_xw_k<false><<<dim3((BB * MM) / 256, 8), 128, 0, stream>>>(f_ae, Pp[pff_W], f_t2, BB * MM);
    gemm_xw_k<false><<<dim3((BB * MM) / 256, 8), 128, 0, stream>>>(f_t1, Pp[pff_W], f_t3, BB * MM);
    k_ista<<<(int)((size_t)BB * CM / 256), 256, 0, stream>>>(f_ae, f_t2, f_t3, f_ff, (int)((size_t)BB * CM));
    k_gn<<<dim3(NGRP, BB), 256, 0, stream>>>(f_ff, Pp[pff_gn_s], Pp[pff_gn_b], DIM_ / NGRP, MM, 1, CM);
    for (int b = 0; b < BB; ++b)
        gemm_k<XD, E_STORE, R_NONE><<<dim3(MM / 256, 8), 128, 0, stream>>>(
            Pp[pW_ffh], Pp[pb_ffh], 1.f, f_ff + b * CM, nullptr, nullptr, nullptr, nullptr,
            nullptr, nullptr, f_creat + b * CM, DIM_, MM, 0, 0, 0, 1.f);

    // ---- geometric layer: pe = W_prenn * sum_k(local_pattern) + K*b ----
    for (int b = 0; b < BB; ++b)
        gemm_k<XD, E_STORE, R_NONE><<<dim3(MM / 256, 8), 128, 0, stream>>>(
            Pp[pW_prenn], Pp[pb_prenn], (float)KK, f_lps + (size_t)b * 4 * MM, nullptr, nullptr,
            nullptr, nullptr, nullptr, nullptr, f_pe + b * CM, 4, MM, 0, 0, 0, 1.f);
    k_sq<<<(BB * MM) / 256, 256, 0, stream>>>(f_pe, f_sq);
    k_graphfeat<<<dim3(MM / 16, BB), 128, 0, stream>>>(f_pe, f_sq, i_gf, f_nrm);

    for (int b = 0; b < BB; ++b) {
        // intra branch: lc1 -> GN/relu -> lc2 with max over K
        gemm_k<XNF, E_STORE, R_NONE><<<dim3(MKs / 256, 8), 128, 0, stream>>>(
            Pp[pW_lc1], Pp[pb_lc1], 1.f, f_pe + b * CM, f_nrm + (size_t)b * MKs, nullptr,
            i_gf + (size_t)b * MKs, nullptr, nullptr, nullptr, f_S1, 2 * DIM_ + 1, (int)MKs, MM, 0, 0, 1.f);
        k_gn<<<dim3(NGRP, 1), 256, 0, stream>>>(f_S1, Pp[pgn_lc_s], Pp[pgn_lc_b], DIM_ / NGRP, (int)MKs, 1, 0);
        gemm_k<XD, E_MAXK, R_NONE><<<dim3(MKs / 256, 8), 128, 0, stream>>>(
            Pp[pW_lc2], Pp[pb_lc2], 1.f, f_S1, nullptr, nullptr, nullptr, nullptr,
            nullptr, nullptr, f_intra + b * CM, DIM_, (int)MKs, 0, 0, MM, 1.f);
        // position encoding: p1 -> GN/relu -> p2
        gemm_k<XPE2, E_STORE, R_NONE><<<dim3(MKs / 256, 8), 128, 0, stream>>>(
            Pp[pW_p1], Pp[pb_p1], 1.f, xyzs + (size_t)b * 3 * NN, out0 + (size_t)b * 3 * MM, nullptr,
            i_knn + (size_t)b * MKs, nullptr, nullptr, nullptr, f_S1, 3, (int)MKs, NN, MM, 0, 1.f);
        k_gn<<<dim3(NGRP, 1), 256, 0, stream>>>(f_S1, Pp[pgn_p_s], Pp[pgn_p_b], DIM_ / NGRP, (int)MKs, 1, 0);
        gemm_k<XD, E_STORE, R_NONE><<<dim3(MKs / 256, 8), 128, 0, stream>>>(
            Pp[pW_p2], Pp[pb_p2], 1.f, f_S1, nullptr, nullptr, nullptr, nullptr,
            nullptr, nullptr, f_S2, DIM_, (int)MKs, 0, 0, 0, 1.f);
        // inter branch: a1(q-key+pos) -> GN/relu -> a2 with masked softmax over K, weighted sum
        gemm_k<XQKP, E_STORE, R_NONE><<<dim3(MKs / 256, 8), 128, 0, stream>>>(
            Pp[pW_a1], Pp[pb_a1], 1.f, f_q + b * CM, f_kv + b * CN, f_S2,
            i_knn + (size_t)b * MKs, nullptr, nullptr, nullptr, f_S1, DIM_, (int)MKs, NN, MM, 0, 1.f);
        k_gn<<<dim3(NGRP, 1), 256, 0, stream>>>(f_S1, Pp[pgn_a_s], Pp[pgn_a_b], DIM_ / NGRP, (int)MKs, 1, 0);
        gemm_k<XD, E_ATTNK, R_NONE><<<dim3(MKs / 256, 8), 128, 0, stream>>>(
            Pp[pW_a2], Pp[pb_a2], 1.f, f_S1, nullptr, f_S2,
            i_knn + (size_t)b * MKs, i_mask + (size_t)b * MKs, f_kv + b * CN, nullptr,
            f_fgt + b * CM, DIM_, (int)MKs, NN, 0, MM, INVSQD);
        // s1
        gemm_k<XD, E_STORE, R_NONE><<<dim3(MM / 256, 8), 128, 0, stream>>>(
            Pp[pW_s1], Pp[pb_s1], 1.f, f_fgt + b * CM, nullptr, nullptr, nullptr, nullptr,
            nullptr, nullptr, f_h3 + b * CM, DIM_, MM, 0, 0, 0, 1.f);
    }
    k_gn<<<dim3(NGRP, BB), 256, 0, stream>>>(f_h3, Pp[pgn_s_s], Pp[pgn_s_b], DIM_ / NGRP, MM, 1, CM);
    for (int b = 0; b < BB; ++b) {
        gemm_k<XD, E_STORE, R_NONE><<<dim3(MM / 256, 8), 128, 0, stream>>>(
            Pp[pW_s2], Pp[pb_s2], 1.f, f_h3 + b * CM, nullptr, nullptr, nullptr, nullptr,
            nullptr, nullptr, f_inter + b * CM, DIM_, MM, 0, 0, 0, 1.f);
        // final aggregation + residual gather of original feats
        gemm_k<XCAT3, E_STORE, R_GATH><<<dim3(MM / 256, 8), 128, 0, stream>>>(
            Pp[pW_post], Pp[pb_post], 1.f, f_creat + b * CM, f_intra + b * CM, f_inter + b * CM,
            nullptr, nullptr, feats + (size_t)b * CN, i_sidx + (size_t)b * MM,
            out1 + b * CM, 3 * DIM_, MM, NN, 0, 0, 1.f);
    }
    (void)in_sizes; (void)out_size; (void)ws_size;
}
